// GCNConv_14826227106020
// MI455X (gfx1250) — matrix operand order repaired and checked by reading, not compiled, not run on hardware
//
#include <hip/hip_runtime.h>

#define D_IN   256
#define D_OUT  256
#define N_SAMP 32
#define TILE_M 16
#define AG_STRIDE 264   // bf16 (ushort) elements per LDS agg row; 264*2=528 bytes, 16B aligned, bank-spread
#define OUT_STRIDE 264  // float elements per LDS out row

typedef __attribute__((ext_vector_type(16))) __bf16 v16bf;
typedef __attribute__((ext_vector_type(8)))  float  v8f;

union Frag16 { unsigned int u[8]; uint4 q[2]; v16bf v; };

__device__ __forceinline__ unsigned short f32_to_bf16_rne(float f) {
    unsigned int u = __float_as_uint(f);
    u += 0x7FFFu + ((u >> 16) & 1u);
    return (unsigned short)(u >> 16);
}

// One-time prep: W [K=256][N=256] f32  ->  Wt [N=256][K=256] bf16 (transposed so
// B-matrix fragments are contiguous per lane -> global_load_b128).
__global__ void __launch_bounds__(256) prep_wt_bf16(const float* __restrict__ W,
                                                    unsigned short* __restrict__ Wt) {
    int idx = blockIdx.x * 256 + threadIdx.x;   // 0..65535
    int k = idx >> 8;
    int n = idx & 255;
    Wt[n * D_IN + k] = f32_to_bf16_rne(W[idx]); // W[idx] == W[k*256+n]
}

__global__ void __launch_bounds__(256)
sage_fused(const int* __restrict__ node_idx,
           const int* __restrict__ neigh_idx,
           const float* __restrict__ features,
           const unsigned short* __restrict__ Wt,   // [256][256] bf16, N-major
           const float* __restrict__ bias,
           float* __restrict__ out,
           int n_nodes)
{
    __shared__ int            s_self[TILE_M];
    __shared__ int            s_neigh[TILE_M * N_SAMP];
    __shared__ unsigned short s_agg[TILE_M * AG_STRIDE];   // bf16 agg tile [16][256]
    __shared__ float          s_out[TILE_M * OUT_STRIDE];  // f32 relu(out) tile [16][256]
    __shared__ float          s_rnorm[TILE_M];

    const int t    = threadIdx.x;
    const int base = blockIdx.x * TILE_M;

    // ---- stage 0: indices into LDS ----
    if (t < TILE_M) {
        int node = base + t;
        s_self[t] = node_idx[node < n_nodes ? node : 0];
    }
    {
        long long lim = (long long)n_nodes * N_SAMP - 1;
        long long i0 = (long long)base * N_SAMP + t;        if (i0 > lim) i0 = lim;
        long long i1 = (long long)base * N_SAMP + 256 + t;  if (i1 > lim) i1 = lim;
        s_neigh[t]       = neigh_idx[i0];
        s_neigh[t + 256] = neigh_idx[i1];
    }
    __syncthreads();

    // ---- stage 1: mean aggregation, float4 per thread ----
    // 64 threads cover one 1KB row (16B each); each quarter of the block owns 4 nodes.
    {
        const int c = (t & 63) * 4;   // feature dims c..c+3
        const int q = t >> 6;         // node subgroup 0..3
        const float inv = 1.0f / 33.0f;
        #pragma unroll 1
        for (int i = 0; i < 4; ++i) {
            const int m = q * 4 + i;
            float4 acc = *(const float4*)&features[(size_t)s_self[m] * D_IN + c];
            #pragma unroll 4
            for (int s = 0; s < N_SAMP; ++s) {
                float4 f = *(const float4*)&features[(size_t)s_neigh[m * N_SAMP + s] * D_IN + c];
                acc.x += f.x; acc.y += f.y; acc.z += f.z; acc.w += f.w;
            }
            uint2 packed;
            packed.x = ((unsigned int)f32_to_bf16_rne(acc.y * inv) << 16) |
                        (unsigned int)f32_to_bf16_rne(acc.x * inv);
            packed.y = ((unsigned int)f32_to_bf16_rne(acc.w * inv) << 16) |
                        (unsigned int)f32_to_bf16_rne(acc.z * inv);
            *(uint2*)&s_agg[m * AG_STRIDE + c] = packed;
        }
    }
    __syncthreads();

    // ---- stage 2: GEMM via v_wmma_f32_16x16x32_bf16 ----
    const int lane = t & 31;
    const int wave = t >> 5;
    const int g    = lane >> 4;   // lane half (ISA lane-group)
    const int mn   = lane & 15;   // A row / B column within tile
    const int n0   = wave * 32;   // this wave owns output cols [n0, n0+32)

    v8f acc0 = {0.f,0.f,0.f,0.f,0.f,0.f,0.f,0.f};
    v8f acc1 = {0.f,0.f,0.f,0.f,0.f,0.f,0.f,0.f};

    const unsigned short* arow  = &s_agg[mn * AG_STRIDE];
    const unsigned short* brow0 = Wt + (size_t)(n0 + mn) * D_IN;
    const unsigned short* brow1 = brow0 + (size_t)16 * D_IN;

    #pragma unroll
    for (int ks = 0; ks < 8; ++ks) {
        Frag16 a, b0, b1;
        // A (16x32), contiguous k order used here (same as B): lanes 0-15 hold K 0..15 contiguous; lanes 16-31 K 16..31.
        a.q[0] = *(const uint4*)(arow + ks * 32 + 16 * g);
        a.q[1] = *(const uint4*)(arow + ks * 32 + 16 * g + 8);
        // B (32x16), contiguous k order used here (same as A): lanes 0-15 hold K 0..15 contiguous; lanes 16-31 K 16..31.
        b0.q[0] = *(const uint4*)(brow0 + ks * 32 + 16 * g);
        b0.q[1] = *(const uint4*)(brow0 + ks * 32 + 16 * g + 8);
        b1.q[0] = *(const uint4*)(brow1 + ks * 32 + 16 * g);
        b1.q[1] = *(const uint4*)(brow1 + ks * 32 + 16 * g + 8);

        acc0 = __builtin_amdgcn_wmma_f32_16x16x32_bf16(false, a.v, false, b0.v,
                                                       (short)0, acc0, false, false);
        acc1 = __builtin_amdgcn_wmma_f32_16x16x32_bf16(false, a.v, false, b1.v,
                                                       (short)0, acc1, false, false);
    }

    // ---- stage 3: bias + ReLU, spill C tiles to LDS (C/D layout: M = j + 8*g, N = mn) ----
    {
        float bb0 = bias[n0 + mn];
        float bb1 = bias[n0 + 16 + mn];
        #pragma unroll
        for (int j = 0; j < 8; ++j) {
            int m = j + 8 * g;
            s_out[m * OUT_STRIDE + n0 + mn]      = fmaxf(acc0[j] + bb0, 0.0f);
            s_out[m * OUT_STRIDE + n0 + 16 + mn] = fmaxf(acc1[j] + bb1, 0.0f);
        }
    }
    __syncthreads();

    // ---- stage 4: per-row L2 norm ----
    if (t < TILE_M) {
        float ssum = 0.0f;
        #pragma unroll 8
        for (int n = 0; n < D_OUT; ++n) {
            float x = s_out[t * OUT_STRIDE + n];
            ssum += x * x;
        }
        s_rnorm[t] = 1.0f / fmaxf(sqrtf(ssum), 1e-12f);
    }
    __syncthreads();

    // ---- stage 5: normalized coalesced store ----
    #pragma unroll 1
    for (int m = 0; m < TILE_M; ++m) {
        int node = base + m;
        if (node < n_nodes)
            out[(size_t)node * D_OUT + t] = s_out[m * OUT_STRIDE + t] * s_rnorm[m];
    }
}

extern "C" void kernel_launch(void* const* d_in, const int* in_sizes, int n_in,
                              void* d_out, int out_size, void* d_ws, size_t ws_size,
                              hipStream_t stream) {
    const int*   node_idx  = (const int*)  d_in[0];
    const int*   neigh_idx = (const int*)  d_in[1];
    const float* features  = (const float*)d_in[2];
    const float* W         = (const float*)d_in[3];
    const float* bias      = (const float*)d_in[4];
    float*       out       = (float*)d_out;

    unsigned short* Wt = (unsigned short*)d_ws;  // 256*256 bf16 = 128 KB

    const int n_nodes = in_sizes[0];             // 50000

    prep_wt_bf16<<<256, 256, 0, stream>>>(W, Wt);

    const int nblocks = (n_nodes + TILE_M - 1) / TILE_M;  // 3125
    sage_fused<<<nblocks, 256, 0, stream>>>(node_idx, neigh_idx, features, Wt,
                                            bias, out, n_nodes);
}
